// Attention_71305047048453
// MI455X (gfx1250) — matrix operand order repaired and checked by reading, not compiled, not run on hardware
//
#include <hip/hip_runtime.h>

typedef __bf16 bf16;
typedef __bf16 v8bf  __attribute__((ext_vector_type(8)));
typedef __bf16 v16bf __attribute__((ext_vector_type(16)));
typedef float  v8f   __attribute__((ext_vector_type(8)));
typedef int    v4i   __attribute__((ext_vector_type(4)));

#define WMMA_BF16 __builtin_amdgcn_wmma_f32_16x16x32_bf16

#if __has_builtin(__builtin_amdgcn_global_load_async_to_lds_b128) && \
    __has_builtin(__builtin_amdgcn_s_wait_asynccnt)
#define HAS_ASYNC 1
#else
#define HAS_ASYNC 0
#endif

#define AS1 __attribute__((address_space(1)))
#define AS3 __attribute__((address_space(3)))

static constexpr int Bc = 2;       // batch
static constexpr int Sc = 2048;    // seq
static constexpr int Dc = 2048;    // model dim
static constexpr int Hc = 16;      // heads
static constexpr int DHc = 128;    // head dim
static constexpr float ATT_SCALE = 0.08838834764831845f; // 1/sqrt(128)
static constexpr float NEGV = -1e9f;

// ---------------------------------------------------------------- f32 -> bf16
__global__ void cvt_f32_to_bf16(const float* __restrict__ src,
                                bf16* __restrict__ dst, int n) {
  int i = blockIdx.x * blockDim.x + threadIdx.x;
  if (i < n) dst[i] = (bf16)src[i];
}

// ---------------------------------------------------------- async B128 copy
__device__ __forceinline__ void copy_b128_to_lds(const bf16* g, bf16* l) {
#if HAS_ASYNC
  __builtin_amdgcn_global_load_async_to_lds_b128(
      (AS1 v4i*)(AS1 void*)g, (AS3 v4i*)(AS3 void*)l, 0, 0);
#else
  *(uint4*)l = *(const uint4*)g;
#endif
}

// ------------------------------------------------- GEMM: C = A * W^T + bias
// A: [M,K] bf16 row-major, W: [N,K] bf16 row-major, C: [M,N] f32.
// Block = 4 waves, block tile 64x64. Weight tile (64 n-rows x 32 k) is
// double-buffered in LDS via async-to-LDS DMA and shared by all 4 waves;
// each wave computes 16x64 (A fragment reused for 4 WMMAs).
__global__ __launch_bounds__(128)
void gemm_bf16_nt(const bf16* __restrict__ A, const bf16* __restrict__ W,
                  const float* __restrict__ bias, float* __restrict__ C,
                  int M, int N, int K) {
  __shared__ __align__(128) bf16 bsh[2][64 * 32];   // [buf][n][k] 4KB each

  const int tid  = threadIdx.x;
  const int wave = tid >> 5;
  const int lane = tid & 31;
  const int l15  = lane & 15;
  const int half = lane >> 4;

  const int ntn = N >> 6;                 // 64-wide n tiles
  const int bm  = blockIdx.x / ntn;
  const int bn  = blockIdx.x % ntn;
  const int m0  = bm * 64 + wave * 16;    // this wave's 16 rows
  const int n0  = bn * 64;                // block's 64 weight rows

  const bf16* __restrict__ arow = A + (size_t)(m0 + l15) * K + half * 16;

  v8f acc[4];
#pragma unroll
  for (int t = 0; t < 4; ++t) acc[t] = (v8f){0,0,0,0,0,0,0,0};

  // stage one 64x32 weight tile: 4KB = 256 x 16B chunks, 128 threads x 2
  auto stage = [&](int buf, int k0) {
#pragma unroll
    for (int rnd = 0; rnd < 2; ++rnd) {
      const int c  = rnd * 128 + tid;
      const int n  = c >> 2;              // tile row
      const int ko = (c & 3) * 8;         // k offset (elements)
      copy_b128_to_lds(W + (size_t)(n0 + n) * K + k0 + ko,
                       &bsh[buf][n * 32 + ko]);
    }
  };

  const int nsteps = K >> 5;
  stage(0, 0);
  int cur = 0;
  for (int s = 0; s < nsteps; ++s) {
    const int k0 = s << 5;
    const bool pre = (s + 1 < nsteps);
    if (pre) stage(cur ^ 1, k0 + 32);
#if HAS_ASYNC
    if (pre) __builtin_amdgcn_s_wait_asynccnt(2);
    else     __builtin_amdgcn_s_wait_asynccnt(0);
#endif
    __syncthreads();                      // current buffer visible to block

    // A fragment: lane holds row m, 16 consecutive k (k = 16*half + s)
    v16bf a = *(const v16bf*)(arow + k0);
    // B fragments from LDS: lane holds col n, 16 consecutive k
#pragma unroll
    for (int t = 0; t < 4; ++t) {
      v16bf b = *(const v16bf*)(&bsh[cur][(t * 16 + l15) * 32 + half * 16]);
      acc[t] = WMMA_BF16(false, a, false, b, (short)0, acc[t], false, false);
    }

    __syncthreads();                      // done reading before next overwrite
    cur ^= 1;
  }

  const int mrow = m0 + half * 8;
#pragma unroll
  for (int t = 0; t < 4; ++t) {
    const int col = n0 + t * 16 + l15;
    const float bv = bias[col];
#pragma unroll
    for (int r = 0; r < 8; ++r)
      C[(size_t)(mrow + r) * N + col] = acc[t][r] + bv;
  }
}

// ------------------------------------- RoPE (half-split variant) + pack bf16
// in : f32 [B,S,H*DH]; freqs f32 [S, DH/2, 2]; out: bf16 [B,H,S,DH]
__global__ void rope_pack(const float* __restrict__ x,
                          const float* __restrict__ fc,
                          bf16* __restrict__ out, int n) {
  int i = blockIdx.x * blockDim.x + threadIdx.x;
  if (i >= n) return;
  const int d = i & (DHc - 1);
  const int h = (i >> 7) & (Hc - 1);
  const int s = (i >> 11) & (Sc - 1);
  const int b = i >> 22;
  const int dd = d & 63;
  const float* xb = x + ((size_t)(b * Sc + s) * Hc + h) * DHc;
  const float cs = fc[(s * 64 + dd) * 2 + 0];
  const float sn = fc[(s * 64 + dd) * 2 + 1];
  const float x0 = xb[2 * dd];
  const float x1 = xb[2 * dd + 1];
  const float v = (d < 64) ? (x0 * cs - x1 * sn) : (x0 * sn + x1 * cs);
  out[((size_t)(b * Hc + h) * Sc + s) * DHc + d] = (bf16)v;
}

// ---------------------------------- V pack transposed: bf16 [B,H,DH,S]
__global__ void pack_v_t(const float* __restrict__ x, bf16* __restrict__ out,
                         int n) {
  int i = blockIdx.x * blockDim.x + threadIdx.x;
  if (i >= n) return;
  const int d = i & (DHc - 1);
  const int h = (i >> 7) & (Hc - 1);
  const int s = (i >> 11) & (Sc - 1);
  const int b = i >> 22;
  const float v = x[((size_t)(b * Sc + s) * Hc + h) * DHc + d];
  out[((size_t)(b * Hc + h) * DHc + d) * Sc + s] = (bf16)v;
}

// ------------------------------------------------ Flash attention, causal
// One wave per (b,h,16-row q tile). kv tiles of 32 keys.
// Q,K: bf16 [B,H,S,DH]; Vt: bf16 [B,H,DH,S]; out: bf16 [B,S,H*DH]
__global__ __launch_bounds__(32)
void attn_causal_fa(const bf16* __restrict__ Qb, const bf16* __restrict__ Kb,
                    const bf16* __restrict__ Vt, bf16* __restrict__ Ob) {
  __shared__ __align__(32) bf16 p_lds[16][32];
  const int lane = threadIdx.x & 31;
  const int l15  = lane & 15;
  const int half = lane >> 4;

  const int qt = blockIdx.x & (Sc / 16 - 1);
  const int bh = blockIdx.x / (Sc / 16);
  const int b  = bh / Hc;
  const int h  = bh % Hc;
  const int qbase = qt * 16;

  const bf16* __restrict__ qhead = Qb + (size_t)bh * Sc * DHc;
  const bf16* __restrict__ khead = Kb + (size_t)bh * Sc * DHc;
  const bf16* __restrict__ vhead = Vt + (size_t)bh * DHc * Sc;

  v16bf qf[4];
  {
    const bf16* qrow = qhead + (size_t)(qbase + l15) * DHc + half * 16;
#pragma unroll
    for (int c = 0; c < 4; ++c) {
      qf[c] = *(const v16bf*)(qrow + c * 32);
    }
  }

  const v8f vzero = {0,0,0,0,0,0,0,0};
  v8f o[8];
#pragma unroll
  for (int t = 0; t < 8; ++t) o[t] = vzero;
  float mrow[8], lrow[8];
#pragma unroll
  for (int r = 0; r < 8; ++r) { mrow[r] = -1e30f; lrow[r] = 0.f; }

  const int kend = qbase + 16;          // causal: keys <= qbase+15
  for (int j = 0; j < kend; j += 32) {
    v8f s0 = vzero, s1 = vzero;
    const bf16* k0p = khead + (size_t)(j + l15) * DHc + half * 16;
    const bf16* k1p = khead + (size_t)(j + 16 + l15) * DHc + half * 16;
#pragma unroll
    for (int c = 0; c < 4; ++c) {
      v16bf kb0 = *(const v16bf*)(k0p + c * 32);
      v16bf kb1 = *(const v16bf*)(k1p + c * 32);
      s0 = WMMA_BF16(false, qf[c], false, kb0, (short)0, s0, false, false);
      s1 = WMMA_BF16(false, qf[c], false, kb1, (short)0, s1, false, false);
    }

    float p0[8], p1[8];
#pragma unroll
    for (int r = 0; r < 8; ++r) {
      const int m  = qbase + half * 8 + r;
      const int n0 = j + l15;
      const int n1 = j + 16 + l15;
      float v0 = s0[r] * ATT_SCALE + (n0 <= m ? 0.f : NEGV);
      float v1 = s1[r] * ATT_SCALE + (n1 <= m ? 0.f : NEGV);
      float mx = fmaxf(v0, v1);
#pragma unroll
      for (int off = 1; off < 16; off <<= 1)
        mx = fmaxf(mx, __shfl_xor(mx, off, 32));
      const float mnew = fmaxf(mrow[r], mx);
      v0 = __expf(v0 - mnew);
      v1 = __expf(v1 - mnew);
      p0[r] = v0; p1[r] = v1;
      float rs = v0 + v1;
#pragma unroll
      for (int off = 1; off < 16; off <<= 1)
        rs += __shfl_xor(rs, off, 32);
      const float f = __expf(mrow[r] - mnew);
      lrow[r] = lrow[r] * f + rs;
      mrow[r] = mnew;
#pragma unroll
      for (int t = 0; t < 8; ++t) o[t][r] *= f;
    }

    // transpose P from C-layout to A-fragment layout through LDS
    __syncthreads();
#pragma unroll
    for (int r = 0; r < 8; ++r) {
      p_lds[half * 8 + r][l15]      = (bf16)p0[r];
      p_lds[half * 8 + r][16 + l15] = (bf16)p1[r];
    }
    __syncthreads();
    v16bf pa = *(const v16bf*)(&p_lds[l15][half * 16]);

#pragma unroll
    for (int t = 0; t < 8; ++t) {
      const bf16* vp = vhead + (size_t)(t * 16 + l15) * Sc + j + half * 16;
      v16bf vb = *(const v16bf*)vp;
      o[t] = WMMA_BF16(false, pa, false, vb, (short)0, o[t], false, false);
    }
  }

#pragma unroll
  for (int r = 0; r < 8; ++r) {
    const float inv = 1.f / lrow[r];
    const int m = qbase + half * 8 + r;
    bf16* orow = Ob + ((size_t)(b * Sc + m) * Dc) + h * DHc;
#pragma unroll
    for (int t = 0; t < 8; ++t)
      orow[t * 16 + l15] = (bf16)(o[t][r] * inv);
  }
}

// ---------------------------------------------------------------- launcher
extern "C" void kernel_launch(void* const* d_in, const int* in_sizes, int n_in,
                              void* d_out, int out_size, void* d_ws,
                              size_t ws_size, hipStream_t stream) {
  const float* hidden = (const float*)d_in[0];
  const float* freqs  = (const float*)d_in[1];
  // d_in[2] = additive causal mask (reproduced analytically in-kernel)
  const float* Wq = (const float*)d_in[3]; const float* bq = (const float*)d_in[4];
  const float* Wk = (const float*)d_in[5]; const float* bk = (const float*)d_in[6];
  const float* Wv = (const float*)d_in[7]; const float* bv = (const float*)d_in[8];
  const float* Wo = (const float*)d_in[9]; const float* bo = (const float*)d_in[10];
  float* out = (float*)d_out;

  const int nHid = Bc * Sc * Dc;      // 8,388,608
  const int nW   = Dc * Dc;           // 4,194,304
  const int M    = Bc * Sc;           // 4096

  char* ws = (char*)d_ws;
  size_t off = 0;
  bf16* hid_bf = (bf16*)(ws + off); off += (size_t)nHid * 2;
  bf16* w_bf   = (bf16*)(ws + off); off += (size_t)nW * 2;
  float* tmp   = (float*)(ws + off); off += (size_t)nHid * 4;
  bf16* q_bf   = (bf16*)(ws + off); off += (size_t)nHid * 2;
  bf16* k_bf   = (bf16*)(ws + off); off += (size_t)nHid * 2;
  bf16* vt_bf  = (bf16*)(ws + off); off += (size_t)nHid * 2;
  bf16* ao_bf  = (bf16*)(ws + off); off += (size_t)nHid * 2;

  const int cvtBlk = 256;
  const dim3 gemmGrid((M / 64) * (Dc / 64));   // 64x64 block tiles
  const dim3 gemmBlk(128);

  cvt_f32_to_bf16<<<nHid / cvtBlk, cvtBlk, 0, stream>>>(hidden, hid_bf, nHid);

  // Q
  cvt_f32_to_bf16<<<nW / cvtBlk, cvtBlk, 0, stream>>>(Wq, w_bf, nW);
  gemm_bf16_nt<<<gemmGrid, gemmBlk, 0, stream>>>(hid_bf, w_bf, bq, tmp, M, Dc, Dc);
  rope_pack<<<nHid / cvtBlk, cvtBlk, 0, stream>>>(tmp, freqs, q_bf, nHid);
  // K
  cvt_f32_to_bf16<<<nW / cvtBlk, cvtBlk, 0, stream>>>(Wk, w_bf, nW);
  gemm_bf16_nt<<<gemmGrid, gemmBlk, 0, stream>>>(hid_bf, w_bf, bk, tmp, M, Dc, Dc);
  rope_pack<<<nHid / cvtBlk, cvtBlk, 0, stream>>>(tmp, freqs, k_bf, nHid);
  // V
  cvt_f32_to_bf16<<<nW / cvtBlk, cvtBlk, 0, stream>>>(Wv, w_bf, nW);
  gemm_bf16_nt<<<gemmGrid, gemmBlk, 0, stream>>>(hid_bf, w_bf, bv, tmp, M, Dc, Dc);
  pack_v_t<<<nHid / cvtBlk, cvtBlk, 0, stream>>>(tmp, vt_bf, nHid);

  // attention: one wave per (b,h,qtile)
  attn_causal_fa<<<Bc * Hc * (Sc / 16), 32, 0, stream>>>(q_bf, k_bf, vt_bf, ao_bf);

  // output projection -> f32 d_out
  cvt_f32_to_bf16<<<nW / cvtBlk, cvtBlk, 0, stream>>>(Wo, w_bf, nW);
  gemm_bf16_nt<<<gemmGrid, gemmBlk, 0, stream>>>(ao_bf, w_bf, bo, out, M, Dc, Dc);
}
